// PRAm_52888227283678
// MI455X (gfx1250) — matrix operand order repaired and checked by reading, not compiled, not run on hardware
//
#include <hip/hip_runtime.h>

// ---------------- problem constants ----------------
#define BB    64
#define SS    333
#define DM    333
#define NH    8
#define DH    64
#define INN   512
#define BS    (BB*SS)      // 21312
#define MPAD  (BS + 64)    // padded M for 128-row block tiles (167*128 = 21376)
#define KPAD  352          // D_MODEL padded to 11*32
#define SQ    336          // q rows padded (21 qtiles * 16)
#define SK    352          // k/v key dim padded (11 * 32)
#define NQKV  1536
#define NWO   384          // Wo N padded to 3*128

typedef __bf16 bf16_t;
typedef __attribute__((ext_vector_type(16))) __bf16 v16bf;
typedef __attribute__((ext_vector_type(8)))  __bf16 v8bf;
typedef __attribute__((ext_vector_type(8)))  float  v8f;

__device__ __forceinline__ bf16_t f2bf(float f) {
  unsigned u = __builtin_bit_cast(unsigned, f);
  u += 0x7FFFu + ((u >> 16) & 1u);
  unsigned short h = (unsigned short)(u >> 16);
  return __builtin_bit_cast(bf16_t, h);
}

union AF { v16bf v; v8bf h[2]; };

// A-matrix 16x32 fragment (16-bit): lanes 0-15 row=lane, K {0..7,16..23};
// lanes 16-31 row=lane-16, K {8..15,24..31}. base points at (row0, k0), stride in halves.
__device__ __forceinline__ v16bf load_afrag(const bf16_t* base, int stride) {
  const int lane = threadIdx.x & 31;
  const int row  = lane & 15;
  const int koff = (lane >> 4) << 3;            // 0 or 8
  const bf16_t* p = base + row * stride + koff;
  AF a;
  a.h[0] = *reinterpret_cast<const v8bf*>(p);
  a.h[1] = *reinterpret_cast<const v8bf*>(p + 16);
  return a.v;
}

// B-matrix 32x16 fragment from an (n, k)-contiguous layout:
// b[e] = B[k = (e<8 ? e : e+8) + 8*(lane>=16)][n = lane&15].
__device__ __forceinline__ v16bf load_bfrag(const bf16_t* base, int stride) {
  const int lane = threadIdx.x & 31;
  const int n    = lane & 15;
  const int koff = (lane >> 4) << 3;            // 0 or 8
  const bf16_t* p = base + n * stride + koff;
  AF b;
  b.h[0] = *reinterpret_cast<const v8bf*>(p);
  b.h[1] = *reinterpret_cast<const v8bf*>(p + 16);
  return b.v;
}

__device__ __forceinline__ v8f wmma_bf16(v16bf a, v16bf b, v8f c) {
  return __builtin_amdgcn_wmma_f32_16x16x32_bf16(false, a, false, b, (short)0, c, false, false);
}

// ---------------- prep kernels ----------------
__global__ __launch_bounds__(256) void embed_kernel(const int* __restrict__ xs,
                                                    const float* __restrict__ emb,
                                                    bf16_t* __restrict__ xbf) {
  long long idx = (long long)blockIdx.x * 256 + threadIdx.x;
  if (idx >= (long long)BS * KPAD) return;
  int row = (int)(idx / KPAD), c = (int)(idx % KPAD);
  float v = 0.f;
  if (c < DM) v = emb[(long long)xs[row] * DM + c];
  xbf[idx] = f2bf(v);
}

// WqkvT[n][k] : (1536 x 352), k-contiguous; n = [Wq | Wk | Wv] columns
__global__ __launch_bounds__(256) void prep_wqkvT(const float* __restrict__ Wq,
                                                  const float* __restrict__ Wk,
                                                  const float* __restrict__ Wv,
                                                  bf16_t* __restrict__ wqkvT) {
  int idx = blockIdx.x * 256 + threadIdx.x;
  if (idx >= NQKV * KPAD) return;
  int n = idx / KPAD, k = idx % KPAD;
  float v = 0.f;
  if (k < DM) {
    int which = n >> 9, cc = n & 511;
    const float* W = (which == 0) ? Wq : (which == 1) ? Wk : Wv;
    v = W[(long long)k * INN + cc];
  }
  wqkvT[idx] = f2bf(v);
}

// WoT[n][k] : (384 x 512), k-contiguous
__global__ __launch_bounds__(256) void prep_woT(const float* __restrict__ Wo,
                                                bf16_t* __restrict__ woT) {
  int idx = blockIdx.x * 256 + threadIdx.x;
  if (idx >= NWO * INN) return;
  int n = idx / INN, k = idx % INN;
  float v = (n < DM) ? Wo[(long long)k * DM + n] : 0.f;
  woT[idx] = f2bf(v);
}

__global__ __launch_bounds__(256) void prep_er(const float* __restrict__ Er,
                                               bf16_t* __restrict__ erbf) {
  int idx = blockIdx.x * 256 + threadIdx.x;
  if (idx >= 352 * DH) return;
  int r = idx / DH, c = idx % DH;
  float v = (r < SS) ? Er[r * DH + c] : 0.f;
  erbf[idx] = f2bf(v);
}

__global__ __launch_bounds__(256) void zero_pads(bf16_t* __restrict__ qbf,
                                                 bf16_t* __restrict__ kbf,
                                                 bf16_t* __restrict__ vt) {
  int idx = blockIdx.x * 256 + threadIdx.x;
  const int QP = 512 * 3 * 64;    // q pad rows 333..335
  const int KP = 512 * 19 * 64;   // k pad rows 333..351
  const int VP = 512 * 64 * 19;   // vt pad cols 333..351
  if (idx < QP) {
    int bh = idx / 192, rem = idx % 192;
    int s = 333 + rem / 64, d = rem % 64;
    qbf[((long long)bh * SQ + s) * DH + d] = f2bf(0.f);
  } else if (idx < QP + KP) {
    int j = idx - QP; int bh = j / 1216, rem = j % 1216;
    int s = 333 + rem / 64, d = rem % 64;
    kbf[((long long)bh * SK + s) * DH + d] = f2bf(0.f);
  } else if (idx < QP + KP + VP) {
    int j = idx - QP - KP; int bh = j / 1216, rem = j % 1216;
    int d = rem / 19, s = 333 + rem % 19;
    vt[((long long)bh * DH + d) * SK + s] = f2bf(0.f);
  }
}

// ---------------- QKV projection GEMM ----------------
// C(21312 x 1536) = Xbf(21312 x 352) @ WqkvT^T, scatter into q/k/vT (bf16).
// 8 waves per block stacked in M (128 rows); each wave owns a 16x128 strip.
__global__ __launch_bounds__(256) void qkv_gemm(const bf16_t* __restrict__ xbf,
                                                const bf16_t* __restrict__ wqkvT,
                                                const float* __restrict__ bq,
                                                const float* __restrict__ bk,
                                                const float* __restrict__ bv,
                                                bf16_t* __restrict__ qbf,
                                                bf16_t* __restrict__ kbf,
                                                bf16_t* __restrict__ vt) {
  const int wave = threadIdx.x >> 5;
  const int lane = threadIdx.x & 31;
  const int Mrow  = blockIdx.x * 128 + wave * 16;
  const int Nbase = blockIdx.y * 128;

  const bf16_t* abase = xbf + (long long)Mrow * KPAD;
  const bf16_t* bbase = wqkvT + (long long)Nbase * KPAD;

  v8f acc[8] = {};
  for (int kk = 0; kk < KPAD; kk += 32) {
    __builtin_prefetch(abase + (lane & 15) * KPAD + kk + 64, 0, 1);
    v16bf af = load_afrag(abase + kk, KPAD);
#pragma unroll
    for (int t = 0; t < 8; ++t) {
      v16bf bf = load_bfrag(bbase + (long long)t * 16 * KPAD + kk, KPAD);
      acc[t] = wmma_bf16(af, bf, acc[t]);
    }
  }

  const int rowoff = (lane >> 4) * 8;
  const int cn = lane & 15;
#pragma unroll
  for (int t = 0; t < 8; ++t) {
#pragma unroll
    for (int r = 0; r < 8; ++r) {
      int m = Mrow + r + rowoff;
      if (m < BS) {
        int n = Nbase + t * 16 + cn;
        int which = n >> 9, hn = n & 511;
        int h = hn >> 6, d = hn & 63;
        int b = m / SS, s = m - b * SS;
        float v = acc[t][r];
        if (which == 0) {
          qbf[((long long)(b * NH + h) * SQ + s) * DH + d] = f2bf(v + bq[hn]);
        } else if (which == 1) {
          kbf[((long long)(b * NH + h) * SK + s) * DH + d] = f2bf(v + bk[hn]);
        } else {
          vt[((long long)(b * NH + h) * DH + d) * SK + s] = f2bf(v + bv[hn]);
        }
      }
    }
  }
}

// ---------------- attention (flash-style, 1 wave per (bh, qtile)) ----------------
#define QE_STRIDE 336
__global__ __launch_bounds__(32) void attn_kernel(const bf16_t* __restrict__ qbf,
                                                  const bf16_t* __restrict__ kbf,
                                                  const bf16_t* __restrict__ vt,
                                                  const bf16_t* __restrict__ erbf,
                                                  bf16_t* __restrict__ obf) {
  __shared__ __align__(16) float qe[16 * QE_STRIDE];
  __shared__ float mrow[16], lrow[16];
  __shared__ __align__(16) bf16_t Pld[16 * 32];

  const int lane = threadIdx.x & 31;
  const int qt = blockIdx.x;
  const int bh = blockIdx.y;
  const int rowoff = (lane >> 4) * 8;
  const int cn = lane & 15;
  const int iBase = qt * 16;

  const bf16_t* qbase = qbf + ((long long)bh * SQ + iBase) * DH;
  v16bf qa0 = load_afrag(qbase, DH);       // K(d) 0..31
  v16bf qa1 = load_afrag(qbase + 32, DH);  // K(d) 32..63

  // relative-position scores: QE = Qtile @ Er^T into LDS
  int er0r = 317 - iBase;
  const int er0 = er0r > 0 ? er0r : 0;
  const int ncols = 333 - er0;
  const int ntile = (ncols + 15) >> 4;
  for (int ct = 0; ct < ntile; ++ct) {
    const bf16_t* eb = erbf + (long long)(er0 + ct * 16) * DH;
    v8f a = {};
    a = wmma_bf16(qa0, load_bfrag(eb, DH), a);
    a = wmma_bf16(qa1, load_bfrag(eb + 32, DH), a);
#pragma unroll
    for (int r = 0; r < 8; ++r)
      qe[(r + rowoff) * QE_STRIDE + ct * 16 + cn] = a[r];
  }
  if (cn == 0) {
#pragma unroll
    for (int r = 0; r < 8; ++r) { mrow[r + rowoff] = -3e38f; lrow[r + rowoff] = 0.f; }
  }
  __syncthreads();

  v8f o0 = {}, o1 = {}, o2 = {}, o3 = {};
  const int kbMax = (iBase + 15) >> 5;
  for (int kb = 0; kb <= kbMax; ++kb) {
    const int K0 = kb * 32;
    v8f sa[2];
#pragma unroll
    for (int sub = 0; sub < 2; ++sub) {
      const bf16_t* kbase = kbf + ((long long)bh * SK + K0 + sub * 16) * DH;
      v8f a = {};
      a = wmma_bf16(qa0, load_bfrag(kbase, DH), a);
      a = wmma_bf16(qa1, load_bfrag(kbase + 32, DH), a);
      sa[sub] = a;
    }
    float pv0[8], pv1[8], tmax[8];
#pragma unroll
    for (int r = 0; r < 8; ++r) {
      const int i = iBase + r + rowoff;
      int j0 = K0 + cn, j1 = K0 + 16 + cn;
      float s0 = (i < SS && j0 <= i)
          ? 0.125f * (sa[0][r] + qe[(r + rowoff) * QE_STRIDE + (332 - i + j0 - er0)]) : -1e30f;
      float s1 = (i < SS && j1 <= i)
          ? 0.125f * (sa[1][r] + qe[(r + rowoff) * QE_STRIDE + (332 - i + j1 - er0)]) : -1e30f;
      pv0[r] = s0; pv1[r] = s1;
      float tm = fmaxf(s0, s1);
#pragma unroll
      for (int off = 8; off; off >>= 1) tm = fmaxf(tm, __shfl_xor(tm, off, 16));
      tmax[r] = tm;
    }
#pragma unroll
    for (int r = 0; r < 8; ++r) {
      const int rr = r + rowoff;
      float mold = mrow[rr];
      float mnew = fmaxf(mold, tmax[r]);
      float scale = __expf(mold - mnew);
      float p0 = __expf(pv0[r] - mnew);
      float p1 = __expf(pv1[r] - mnew);
      float ts = p0 + p1;
#pragma unroll
      for (int off = 8; off; off >>= 1) ts += __shfl_xor(ts, off, 16);
      if (cn == 0) { mrow[rr] = mnew; lrow[rr] = lrow[rr] * scale + ts; }
      o0[r] *= scale; o1[r] *= scale; o2[r] *= scale; o3[r] *= scale;
      Pld[rr * 32 + cn]      = f2bf(p0);
      Pld[rr * 32 + 16 + cn] = f2bf(p1);
    }
    __syncthreads();
    v16bf pf = load_afrag(Pld, 32);
    const bf16_t* vb = vt + (long long)bh * DH * SK + K0;
    o0 = wmma_bf16(pf, load_bfrag(vb + (long long)0  * SK, SK), o0);
    o1 = wmma_bf16(pf, load_bfrag(vb + (long long)16 * SK, SK), o1);
    o2 = wmma_bf16(pf, load_bfrag(vb + (long long)32 * SK, SK), o2);
    o3 = wmma_bf16(pf, load_bfrag(vb + (long long)48 * SK, SK), o3);
    __syncthreads();
  }

  const int b = bh >> 3, h = bh & 7;
#pragma unroll
  for (int r = 0; r < 8; ++r) {
    const int i = iBase + r + rowoff;
    if (i < SS) {
      float l = lrow[r + rowoff];
      float inv = (l > 0.f) ? 1.0f / l : 0.f;
      long long orow = ((long long)b * SS + i) * INN + h * DH;
      obf[orow + 0  + cn] = f2bf(o0[r] * inv);
      obf[orow + 16 + cn] = f2bf(o1[r] * inv);
      obf[orow + 32 + cn] = f2bf(o2[r] * inv);
      obf[orow + 48 + cn] = f2bf(o3[r] * inv);
    }
  }
}

// ---------------- output projection GEMM ----------------
// out(21312 x 333) = Obf(21312 x 512) @ WoT^T + bo
__global__ __launch_bounds__(256) void out_gemm(const bf16_t* __restrict__ obf,
                                                const bf16_t* __restrict__ woT,
                                                const float* __restrict__ bo,
                                                float* __restrict__ out) {
  const int wave = threadIdx.x >> 5;
  const int lane = threadIdx.x & 31;
  const int Mrow  = blockIdx.x * 128 + wave * 16;
  const int Nbase = blockIdx.y * 128;

  const bf16_t* abase = obf + (long long)Mrow * INN;
  const bf16_t* bbase = woT + (long long)Nbase * INN;

  v8f acc[8] = {};
  for (int kk = 0; kk < INN; kk += 32) {
    __builtin_prefetch(abase + (lane & 15) * INN + kk + 64, 0, 1);
    v16bf af = load_afrag(abase + kk, INN);
#pragma unroll
    for (int t = 0; t < 8; ++t) {
      v16bf bf = load_bfrag(bbase + (long long)t * 16 * INN + kk, INN);
      acc[t] = wmma_bf16(af, bf, acc[t]);
    }
  }

  const int rowoff = (lane >> 4) * 8;
  const int cn = lane & 15;
#pragma unroll
  for (int t = 0; t < 8; ++t) {
#pragma unroll
    for (int r = 0; r < 8; ++r) {
      int m = Mrow + r + rowoff;
      int n = Nbase + t * 16 + cn;
      if (m < BS && n < DM) out[(long long)m * DM + n] = acc[t][r] + bo[n];
    }
  }
}

// ---------------- host launcher ----------------
extern "C" void kernel_launch(void* const* d_in, const int* in_sizes, int n_in,
                              void* d_out, int out_size, void* d_ws, size_t ws_size,
                              hipStream_t stream) {
  const int*   xs  = (const int*)  d_in[0];
  const float* emb = (const float*)d_in[1];
  const float* Wq  = (const float*)d_in[2];
  const float* bq  = (const float*)d_in[3];
  const float* Wk  = (const float*)d_in[4];
  const float* bk  = (const float*)d_in[5];
  const float* Wv  = (const float*)d_in[6];
  const float* bv  = (const float*)d_in[7];
  const float* Er  = (const float*)d_in[8];
  const float* Wo  = (const float*)d_in[9];
  const float* bo  = (const float*)d_in[10];
  float* out = (float*)d_out;

  char* ws = (char*)d_ws;
  size_t off = 0;
  auto alloc = [&](size_t bytes) {
    void* p = ws + off;
    off += (bytes + 255) & ~(size_t)255;
    return p;
  };
  bf16_t* xbf   = (bf16_t*)alloc((size_t)MPAD * KPAD * 2);  // padded M rows
  bf16_t* wqkvT = (bf16_t*)alloc((size_t)NQKV * KPAD * 2);
  bf16_t* erbf  = (bf16_t*)alloc((size_t)352 * DH * 2);
  bf16_t* woT   = (bf16_t*)alloc((size_t)NWO * INN * 2);
  bf16_t* qbf   = (bf16_t*)alloc((size_t)512 * SQ * DH * 2);
  bf16_t* kbf   = (bf16_t*)alloc((size_t)512 * SK * DH * 2);
  bf16_t* vt    = (bf16_t*)alloc((size_t)512 * DH * SK * 2);
  bf16_t* obf   = (bf16_t*)alloc((size_t)MPAD * INN * 2);   // padded M rows

  {
    long long tot = (long long)BS * KPAD;
    embed_kernel<<<dim3((unsigned)((tot + 255) / 256)), dim3(256), 0, stream>>>(xs, emb, xbf);
  }
  prep_wqkvT<<<dim3((NQKV * KPAD + 255) / 256), dim3(256), 0, stream>>>(Wq, Wk, Wv, wqkvT);
  prep_er<<<dim3((352 * DH + 255) / 256), dim3(256), 0, stream>>>(Er, erbf);
  prep_woT<<<dim3((NWO * INN + 255) / 256), dim3(256), 0, stream>>>(Wo, woT);
  {
    int tot = 512 * 3 * 64 + 512 * 19 * 64 + 512 * 64 * 19;
    zero_pads<<<dim3((tot + 255) / 256), dim3(256), 0, stream>>>(qbf, kbf, vt);
  }
  qkv_gemm<<<dim3((BS + 127) / 128, NQKV / 128), dim3(256), 0, stream>>>(xbf, wqkvT, bq, bk, bv,
                                                                         qbf, kbf, vt);
  attn_kernel<<<dim3(SQ / 16, BB * NH), dim3(32), 0, stream>>>(qbf, kbf, vt, erbf, obf);
  out_gemm<<<dim3((BS + 127) / 128, NWO / 128), dim3(256), 0, stream>>>(obf, woT, bo, out);
}
